// LithographySimulator_39582418600265
// MI455X (gfx1250) — matrix operand order repaired and checked by reading, not compiled, not run on hardware
//
#include <hip/hip_runtime.h>
#include <hip/hip_bf16.h>
#include <math.h>

typedef __attribute__((ext_vector_type(16))) _Float16 v16h;
typedef __attribute__((ext_vector_type(8)))  float    v8f;

#define Bn   4
#define Hn   1024
#define Wn   1024
#define Kn   24
#define FK   35       // SOCS kernel frequency extent (35x35 window)
#define NT   64       // 1024/16 tiles per dim
#define HWn  (Hn*Wn)

// ang = -2*pi*(f-18)*(n-512)/1024  ->  A[f,n] = exp(i*ang) = forward DFT factor
__device__ __forceinline__ float ang_of(int f, int n) {
    return (-6.283185307179586f / 1024.0f) * (float)((f - 18) * (n - 512));
}

// ---------------------------------------------------------------------------
// K1: twiddle tables (f32) + A-fragments of E (f16, WMMA 16x32 A layout)
//   A-layout (16-bit A, 16x32): lane L: M = L%16, g = L/16;
//   element e in v16h: K = e + 8g (e<8), K = e + 8 + 8g (e>=8)
// ---------------------------------------------------------------------------
__global__ void litho_build_tables(float* __restrict__ Acos, float* __restrict__ Asin,
                                   _Float16* __restrict__ Afr, _Float16* __restrict__ Afi) {
    int tid = blockIdx.x * blockDim.x + threadIdx.x;
    int stride = gridDim.x * blockDim.x;

    for (int i = tid; i < FK * Wn; i += stride) {
        int f = i / Wn, n = i - f * Wn;
        float s, c;
        __sincosf(ang_of(f, n), &s, &c);
        Acos[i] = c;
        Asin[i] = s;
    }
    // E[y,f] = conj(A[f,y]) = (cos, -sin); pad K dim 35 -> 64 with zeros
    for (int i = tid; i < NT * 2 * 32 * 16; i += stride) {
        int e    = i & 15;
        int lane = (i >> 4) & 31;
        int c    = (i >> 9) & 1;     // K chunk: f base = 32*c
        int ty   = i >> 10;          // y tile
        int m = lane & 15, g = lane >> 4;
        int kk = (e < 8) ? (e + 8 * g) : (e + 8 + 8 * g);
        int f  = kk + 32 * c;
        int y  = ty * 16 + m;
        float er = 0.0f, ei = 0.0f;
        if (f < FK) {
            float s, cc;
            __sincosf(ang_of(f, y), &s, &cc);
            er = cc; ei = -s;
        }
        Afr[i] = (_Float16)er;
        Afi[i] = (_Float16)ei;
    }
}

// ---------------------------------------------------------------------------
// K2: T[b,f,x] = sum_y A[f,y] * mask[b,y,x]   (complex, A applied on rows)
// ---------------------------------------------------------------------------
__global__ void litho_stage1(const float* __restrict__ mask,
                             const float* __restrict__ Acos, const float* __restrict__ Asin,
                             float* __restrict__ Tr, float* __restrict__ Ti) {
    int tid = blockIdx.x * blockDim.x + threadIdx.x;
    int stride = gridDim.x * blockDim.x;
    const int total = Bn * FK * Wn;
    for (int i = tid; i < total; i += stride) {
        int x = i & (Wn - 1);
        int f = (i / Wn) % FK;
        int b = i / (Wn * FK);
        const float* mp = mask + (size_t)b * HWn + x;
        const float* ac = Acos + f * Wn;
        const float* as = Asin + f * Wn;
        float sr = 0.0f, si = 0.0f;
        for (int y = 0; y < Hn; ++y) {
            float m = mp[(size_t)y * Wn];      // DOSE = 1.0
            sr = fmaf(ac[y], m, sr);
            si = fmaf(as[y], m, si);
        }
        Tr[i] = sr;
        Ti[i] = si;
    }
}

// ---------------------------------------------------------------------------
// K3: Mhat[b,fy,fx] = sum_x T[b,fy,x] * A[fx,x]   (complex x complex)
// ---------------------------------------------------------------------------
__global__ void litho_stage2(const float* __restrict__ Tr, const float* __restrict__ Ti,
                             const float* __restrict__ Acos, const float* __restrict__ Asin,
                             float* __restrict__ Mr, float* __restrict__ Mi) {
    int tid = blockIdx.x * blockDim.x + threadIdx.x;
    int stride = gridDim.x * blockDim.x;
    const int total = Bn * FK * FK;
    for (int i = tid; i < total; i += stride) {
        int fx = i % FK;
        int fy = (i / FK) % FK;
        int b  = i / (FK * FK);
        const float* tr = Tr + ((size_t)b * FK + fy) * Wn;
        const float* ti = Ti + ((size_t)b * FK + fy) * Wn;
        const float* ac = Acos + fx * Wn;
        const float* as = Asin + fx * Wn;
        float mr = 0.0f, mi = 0.0f;
        for (int x = 0; x < Wn; ++x) {
            float trv = tr[x], tiv = ti[x];
            float acv = ac[x], asv = as[x];
            mr = fmaf(trv, acv, fmaf(-tiv, asv, mr));
            mi = fmaf(trv, asv, fmaf( tiv, acv, mi));
        }
        Mr[i] = mr;
        Mi[i] = mi;
    }
}

// ---------------------------------------------------------------------------
// K4: U[b,k][f,x] = (1/N^2) * sum_fx (Mhat[b]⊙ker[k])[f,fx] * E[x,fx]
//     written directly in WMMA B-fragment layout (16-bit B, 32x16):
//       lane L: N = L%16, g = L/16; element e: K = e + 8g (e<8), K = e + 8 + 8g (e>=8) (+32c chunk)
//     index = ((((b*Kn + k)*NT + tx)*2 + c)*32 + lane)*16 + e
// ---------------------------------------------------------------------------
__global__ void litho_stage3(const float* __restrict__ Mr, const float* __restrict__ Mi,
                             const float* __restrict__ kern /* complex64 [K][35][35] */,
                             const float* __restrict__ Acos, const float* __restrict__ Asin,
                             _Float16* __restrict__ Ufr, _Float16* __restrict__ Ufi) {
    const float invN2 = 1.0f / (1024.0f * 1024.0f);
    int tid = blockIdx.x * blockDim.x + threadIdx.x;
    int stride = gridDim.x * blockDim.x;
    const int total = Bn * Kn * NT * 2 * 32 * 16;   // 6,291,456
    for (int i = tid; i < total; i += stride) {
        int e    = i & 15;
        int lane = (i >> 4) & 31;
        int c    = (i >> 9) & 1;
        int tx   = (i >> 10) & (NT - 1);
        int bk   = i >> 16;                 // b*Kn + k (65536 entries per (b,k))
        int k    = bk % Kn;
        int b    = bk / Kn;
        int n = lane & 15, g = lane >> 4;
        int f = ((e < 8) ? (e + 8 * g) : (e + 8 + 8 * g)) + 32 * c;        // padded K index
        int x = tx * 16 + n;
        float ur = 0.0f, ui = 0.0f;
        if (f < FK) {
            const float* mr = Mr + ((size_t)b * FK + f) * FK;
            const float* mi = Mi + ((size_t)b * FK + f) * FK;
            const float* kp = kern + ((size_t)(k * FK + f) * FK) * 2;
            for (int fx = 0; fx < FK; ++fx) {
                float mrv = mr[fx], miv = mi[fx];
                float krv = kp[2 * fx], kiv = kp[2 * fx + 1];
                float srv = mrv * krv - miv * kiv;   // S = Mhat * ker
                float siv = mrv * kiv + miv * krv;
                float ec = Acos[fx * Wn + x];
                float es = Asin[fx * Wn + x];
                // E[x,fx] = (ec, -es):  (sr+i si)(ec - i es)
                ur = fmaf(srv, ec, fmaf( siv, es, ur));
                ui = fmaf(siv, ec, fmaf(-srv, es, ui));
            }
        }
        Ufr[i] = (_Float16)(ur * invN2);
        Ufi[i] = (_Float16)(ui * invN2);
    }
}

// ---------------------------------------------------------------------------
// K5: field = E @ U (complex GEMM via 4 real WMMAs x 2 K-chunks), then
//     aerial += scale_k * |field|^2 ; resist = sigmoid ; printed = >0.5
//     One wave (32 lanes) per 16x16 output tile.
// ---------------------------------------------------------------------------
__global__ void __launch_bounds__(256)
litho_aerial_wmma(const _Float16* __restrict__ Afr, const _Float16* __restrict__ Afi,
                  const _Float16* __restrict__ Ufr, const _Float16* __restrict__ Ufi,
                  const float* __restrict__ scales, float* __restrict__ out) {
    int wave = (blockIdx.x * blockDim.x + threadIdx.x) >> 5;
    int lane = threadIdx.x & 31;
    if (wave >= Bn * NT * NT) return;
    int tx = wave & (NT - 1);
    int ty = (wave >> 6) & (NT - 1);
    int b  = wave >> 12;

    // A-fragments of E for this y-tile (shared by all k and all x-tiles)
    const size_t abase = ((size_t)ty * 2 * 32 + lane) * 16;
    v16h aEr0 = *(const v16h*)(Afr + abase);
    v16h aEr1 = *(const v16h*)(Afr + abase + 512);
    v16h aEi0 = *(const v16h*)(Afi + abase);
    v16h aEi1 = *(const v16h*)(Afi + abase + 512);

    float acc[8];
#pragma unroll
    for (int r = 0; r < 8; ++r) acc[r] = 0.0f;

    for (int k = 0; k < Kn; ++k) {
        const size_t ub = (((size_t)((b * Kn + k) * NT + tx) * 2) * 32 + lane) * 16;
        v16h bUr0 = *(const v16h*)(Ufr + ub);
        v16h bUr1 = *(const v16h*)(Ufr + ub + 512);
        v16h bUi0 = *(const v16h*)(Ufi + ub);
        v16h bUi1 = *(const v16h*)(Ufi + ub + 512);

        v8f z = {};
        v8f crr = __builtin_amdgcn_wmma_f32_16x16x32_f16(false, aEr0, false, bUr0, (short)0, z,   false, false);
        crr     = __builtin_amdgcn_wmma_f32_16x16x32_f16(false, aEr1, false, bUr1, (short)0, crr, false, false);
        v8f cii = __builtin_amdgcn_wmma_f32_16x16x32_f16(false, aEi0, false, bUi0, (short)0, z,   false, false);
        cii     = __builtin_amdgcn_wmma_f32_16x16x32_f16(false, aEi1, false, bUi1, (short)0, cii, false, false);
        v8f cri = __builtin_amdgcn_wmma_f32_16x16x32_f16(false, aEr0, false, bUi0, (short)0, z,   false, false);
        cri     = __builtin_amdgcn_wmma_f32_16x16x32_f16(false, aEr1, false, bUi1, (short)0, cri, false, false);
        v8f cir = __builtin_amdgcn_wmma_f32_16x16x32_f16(false, aEi0, false, bUr0, (short)0, z,   false, false);
        cir     = __builtin_amdgcn_wmma_f32_16x16x32_f16(false, aEi1, false, bUr1, (short)0, cir, false, false);

        float s = scales[k];
#pragma unroll
        for (int r = 0; r < 8; ++r) {
            float re = crr[r] - cii[r];
            float im = cri[r] + cir[r];
            acc[r] = fmaf(s, fmaf(re, re, im * im), acc[r]);
        }
    }

    // C/D layout: vgpr r -> M = r + 8*(lane/16), N = lane%16
    int g = lane >> 4, n = lane & 15;
    int x = tx * 16 + n;
    const size_t BHW = (size_t)Bn * HWn;
#pragma unroll
    for (int r = 0; r < 8; ++r) {
        int y = ty * 16 + r + 8 * g;
        size_t idx = (size_t)b * HWn + (size_t)y * Wn + x;
        float a = acc[r];
        float resist = 1.0f / (1.0f + __expf(-50.0f * (a - 0.225f)));
        float printed = (resist > 0.5f) ? 1.0f : 0.0f;
        out[idx]           = a;
        out[BHW + idx]     = resist;
        out[2 * BHW + idx] = printed;
    }
}

// ---------------------------------------------------------------------------
static inline size_t align256(size_t v) { return (v + 255) & ~(size_t)255; }

extern "C" void kernel_launch(void* const* d_in, const int* in_sizes, int n_in,
                              void* d_out, int out_size, void* d_ws, size_t ws_size,
                              hipStream_t stream) {
    const float* mask   = (const float*)d_in[0];                 // [4,1024,1024] f32
    const float* kern   = (const float*)d_in[1];                 // [24,35,35] complex64
    const float* scales = (const float*)d_in[3];                 // [24] f32
    float* out = (float*)d_out;                                  // aerial|resist|printed

    // workspace carve-up
    char* ws = (char*)d_ws;
    size_t off = 0;
    float*    Acos = (float*)(ws + off); off = align256(off + (size_t)FK * Wn * 4);
    float*    Asin = (float*)(ws + off); off = align256(off + (size_t)FK * Wn * 4);
    float*    Tr   = (float*)(ws + off); off = align256(off + (size_t)Bn * FK * Wn * 4);
    float*    Ti   = (float*)(ws + off); off = align256(off + (size_t)Bn * FK * Wn * 4);
    float*    Mr   = (float*)(ws + off); off = align256(off + (size_t)Bn * FK * FK * 4);
    float*    Mi   = (float*)(ws + off); off = align256(off + (size_t)Bn * FK * FK * 4);
    _Float16* Afr  = (_Float16*)(ws + off); off = align256(off + (size_t)NT * 2 * 32 * 16 * 2);
    _Float16* Afi  = (_Float16*)(ws + off); off = align256(off + (size_t)NT * 2 * 32 * 16 * 2);
    _Float16* Ufr  = (_Float16*)(ws + off); off = align256(off + (size_t)Bn * Kn * NT * 2 * 32 * 16 * 2);
    _Float16* Ufi  = (_Float16*)(ws + off); off = align256(off + (size_t)Bn * Kn * NT * 2 * 32 * 16 * 2);
    (void)ws_size; (void)in_sizes; (void)n_in; (void)out_size;

    // K1: tables + E A-fragments
    litho_build_tables<<<256, 256, 0, stream>>>(Acos, Asin, Afr, Afi);

    // K2: T = A @ mask
    {
        int total = Bn * FK * Wn;
        litho_stage1<<<(total + 255) / 256, 256, 0, stream>>>(mask, Acos, Asin, Tr, Ti);
    }
    // K3: Mhat = T @ A^T
    {
        int total = Bn * FK * FK;
        litho_stage2<<<(total + 255) / 256, 256, 0, stream>>>(Tr, Ti, Acos, Asin, Mr, Mi);
    }
    // K4: U fragments (B layout)
    {
        int total = Bn * Kn * NT * 2 * 32 * 16;
        litho_stage3<<<(total + 255) / 256, 256, 0, stream>>>(Mr, Mi, kern, Acos, Asin, Ufr, Ufi);
    }
    // K5: WMMA complex GEMM + intensity + resist + printed
    {
        int waves  = Bn * NT * NT;              // 16384 tiles, 1 wave each
        int blocks = (waves + 7) / 8;           // 8 waves / 256-thread block
        litho_aerial_wmma<<<blocks, 256, 0, stream>>>(Afr, Afi, Ufr, Ufi, scales, out);
    }
}
